// SparseAttention_61624190763283
// MI455X (gfx1250) — compile-verified, matrix operand order checked by reading, not run on hardware
//
#include <hip/hip_runtime.h>

// Problem constants (from reference): B=1, N=50000, H=8, E=D=32, P=1,600,000
#define N_NODES   50000
#define N_HEADS   8
#define EDIM      32
#define N_EDGES   1600000
#define TEMP      0.17677669529663689f   // 1/sqrt(32)

typedef __attribute__((ext_vector_type(16))) _Float16 v16h;
typedef __attribute__((ext_vector_type(8)))  float    v8f;

// ---- monotone float <-> uint key so segment-max is a single atomicMax(u32) ----
__device__ __forceinline__ unsigned fkey(float f) {
  unsigned u = __float_as_uint(f);
  return (u & 0x80000000u) ? ~u : (u | 0x80000000u);
}
__device__ __forceinline__ float keyf(unsigned kk) {
  unsigned u = (kk & 0x80000000u) ? (kk ^ 0x80000000u) : ~kk;
  return __uint_as_float(u);
}

// runtime-index read of one element of the WMMA f32 accumulator (cndmask chain)
__device__ __forceinline__ float v8f_get(v8f c, int r) {
  float x = c[0];
  if (r == 1) x = c[1];
  if (r == 2) x = c[2];
  if (r == 3) x = c[3];
  if (r == 4) x = c[4];
  if (r == 5) x = c[5];
  if (r == 6) x = c[6];
  if (r == 7) x = c[7];
  return x;
}

// -----------------------------------------------------------------------------
// Pass 1: per-edge scores via v_wmma_f32_16x16x32_f16.
// One block = one tile of 16 edges; each of the 8 waves handles one head.
// A row e  = q[src_e, h, 0:32] (f32 gathered, cvt to f16)
// B col e  = k[dst_e, h, 0:32]
// diag(AxB) = 16 edge dot products. Diagonal (e,e) lives in:
//   e in 0..7  -> lane e,     accum VGPR e
//   e in 8..15 -> lane e+16,  accum VGPR e-8
// and for those lanes (lane&15)==e, so the lane already holds src_e.
// -----------------------------------------------------------------------------
__global__ __launch_bounds__(256) void gat_scores_wmma(
    const float* __restrict__ q, const float* __restrict__ k,
    const int* __restrict__ adj, float* __restrict__ sbuf,
    unsigned* __restrict__ mkey) {
  const int lane = threadIdx.x & 31;
  const int h    = threadIdx.x >> 5;        // wave-in-block = head
  const int tile = blockIdx.x;
  const int eloc = lane & 15;
  const int edge = tile * 16 + eloc;

  const int srcn = adj[edge];               // adj[0][e]
  const int dstn = adj[N_EDGES + edge];     // adj[1][e]

  // ISA 16-bit A 16x32 layout: lanes 0-15 carry K 0..7 & 16..23,
  // lanes 16-31 carry K 8..15 & 24..31. B mirrors A (column-major).
  const int kb = (lane < 16) ? 0 : 8;
  const float* qrow = q + ((size_t)srcn * N_HEADS + h) * EDIM + kb;
  const float* krow = k + ((size_t)dstn * N_HEADS + h) * EDIM + kb;

  v16h A, Bm;
#pragma unroll
  for (int i = 0; i < 8; ++i) {
    A[i]      = (_Float16)qrow[i];
    A[i + 8]  = (_Float16)qrow[i + 16];
    Bm[i]     = (_Float16)krow[i];
    Bm[i + 8] = (_Float16)krow[i + 16];
  }

  v8f C = {};
  C = __builtin_amdgcn_wmma_f32_16x16x32_f16(
      /*neg_a=*/false, A, /*neg_b=*/false, Bm,
      /*c_mod=*/(short)0, C, /*reuse_a=*/false, /*reuse_b=*/false);

  const bool active = (lane < 8) || (lane >= 24);
  if (active) {
    const int r = (lane < 8) ? lane : (lane - 24);   // accumulator VGPR index
    const float s = v8f_get(C, r) * TEMP;            // edge = tile*16 + (lane&15)
    sbuf[(size_t)edge * N_HEADS + h] = s;
    atomicMax(mkey + (size_t)srcn * N_HEADS + h, fkey(s));
  }
}

// -----------------------------------------------------------------------------
// Pass 2: p = exp(s - m[src]) in-place; atomicAdd denominator.
// 8 consecutive threads = 8 heads of one edge -> contiguous 32B atomic targets.
// -----------------------------------------------------------------------------
__global__ __launch_bounds__(256) void gat_softmax_norm(
    const int* __restrict__ adj, const unsigned* __restrict__ mkey,
    float* __restrict__ sbuf, float* __restrict__ den) {
  const size_t t = (size_t)blockIdx.x * blockDim.x + threadIdx.x;
  const int e = (int)(t >> 3);
  const int h = (int)(t & 7);
  const int srcn = adj[e];
  const float m = keyf(mkey[(size_t)srcn * N_HEADS + h]);
  const float p = __expf(sbuf[t] - m);
  sbuf[t] = p;
  atomicAdd(den + (size_t)srcn * N_HEADS + h, p);
}

// -----------------------------------------------------------------------------
// Pass 3: out[src,h,:] += (p/den) * v[dst,h,:].
// One block per edge; thread = (head, d). v gather is a 128B coalesced
// global_load per wave; scatter is global_atomic_add_f32 (L2-resident).
// -----------------------------------------------------------------------------
__global__ __launch_bounds__(256) void gat_aggregate(
    const int* __restrict__ adj, const float* __restrict__ v,
    const float* __restrict__ sbuf, const float* __restrict__ den,
    float* __restrict__ out) {
  const int e = blockIdx.x;
  const int h = threadIdx.x >> 5;
  const int d = threadIdx.x & 31;
  const int srcn = adj[e];
  const int dstn = adj[N_EDGES + e];
  const float p  = sbuf[(size_t)e * N_HEADS + h];
  const float dn = den[(size_t)srcn * N_HEADS + h];
  const float alpha = p / dn;
  const float vv = v[((size_t)dstn * N_HEADS + h) * EDIM + d];
  atomicAdd(out + ((size_t)srcn * N_HEADS + h) * EDIM + d, alpha * vv);
}

extern "C" void kernel_launch(void* const* d_in, const int* in_sizes, int n_in,
                              void* d_out, int out_size, void* d_ws, size_t ws_size,
                              hipStream_t stream) {
  const float* q   = (const float*)d_in[0];   // (1, N, H, E) f32
  const float* k   = (const float*)d_in[1];   // (1, N, H, E) f32
  const float* v   = (const float*)d_in[2];   // (1, N, H, E) f32
  const int*   adj = (const int*)  d_in[3];   // (2, P) i32
  float* out = (float*)d_out;                 // (1, N, H, E) f32

  // workspace layout: [ sbuf: P*H f32 | mkey: N*H u32 | den: N*H f32 ]
  float*    sbuf = (float*)d_ws;
  unsigned* mkey = (unsigned*)((char*)d_ws + (size_t)N_EDGES * N_HEADS * sizeof(float));
  float*    den  = (float*)((char*)mkey + (size_t)N_NODES * N_HEADS * sizeof(unsigned));

  // per-call init (graph-capture-safe memset nodes). mkey=0 is the minimum key.
  hipMemsetAsync(out,  0, (size_t)N_NODES * N_HEADS * EDIM * sizeof(float), stream);
  hipMemsetAsync(mkey, 0, (size_t)N_NODES * N_HEADS * sizeof(unsigned), stream);
  hipMemsetAsync(den,  0, (size_t)N_NODES * N_HEADS * sizeof(float), stream);

  gat_scores_wmma<<<N_EDGES / 16, 256, 0, stream>>>(q, k, adj, sbuf, mkey);
  gat_softmax_norm<<<(N_EDGES * N_HEADS) / 256, 256, 0, stream>>>(adj, mkey, sbuf, den);
  gat_aggregate<<<N_EDGES, 256, 0, stream>>>(adj, v, sbuf, den, out);
}
